// TGCNModel_89739046682923
// MI455X (gfx1250) — matrix operand order repaired and checked by reading, not compiled, not run on hardware
//
#include <hip/hip_runtime.h>
#include <hip/hip_bf16.h>
#include <math.h>

typedef __attribute__((ext_vector_type(16))) _Float16 v16h;
typedef __attribute__((ext_vector_type(8)))  float    v8f;

#define FDIM   512
#define LDSTR  48   // LDS row stride in halves (96B, multiple of 16B, bank-spread)

// ---------------------------------------------------------------------------
// Degree / normalization kernels
// ---------------------------------------------------------------------------
__global__ void deg_init_kernel(float* deg, int n) {
  int i = blockIdx.x * blockDim.x + threadIdx.x;
  if (i < n) deg[i] = 1.0f;  // self-loop weight
}

__global__ void deg_edges_kernel(const int* __restrict__ dst,
                                 const float* __restrict__ ew,
                                 float* deg, int e) {
  int i = blockIdx.x * blockDim.x + threadIdx.x;
  if (i < e) atomicAdd(&deg[dst[i]], ew[i]);
}

__global__ void deg_fin_kernel(float* deg, int n) {
  int i = blockIdx.x * blockDim.x + threadIdx.x;
  if (i < n) {
    float d = deg[i];
    deg[i] = (d > 0.0f) ? rsqrtf(d) : 0.0f;   // deg -> dinv in place
  }
}

// ---------------------------------------------------------------------------
// GEMM: C[M x 512] = A[M x 512] @ B[512 x 512], f32 in/out, f16 WMMA math.
// Block tile 128x128, 256 threads = 8 waves (2 x 4), each wave 64x32 = 8 WMMA
// accumulators. K staged through LDS in 32-wide slabs (one wmma K per slab).
// blockIdx.x = column tile (fast dim -> A-tile L2 reuse), blockIdx.y = row tile.
// ---------------------------------------------------------------------------
__global__ __launch_bounds__(256)
void gemm_wmma_f16(const float* __restrict__ A, const float* __restrict__ B,
                   float* __restrict__ C, int M) {
  __shared__ _Float16 Alds[128 * LDSTR];   // [row 0..127][k 0..31]
  __shared__ _Float16 Blds[128 * LDSTR];   // transposed: [n 0..127][k 0..31]

  const int tid   = threadIdx.x;
  const int lane  = tid & 31;
  const int wave  = tid >> 5;
  const int waveM = wave >> 2;     // 0..1  -> 64 rows each
  const int waveN = wave & 3;      // 0..3  -> 32 cols each
  const int l16   = lane & 15;
  const int hi    = lane >> 4;     // lane half select
  const int row0  = blockIdx.y * 128;
  const int col0  = blockIdx.x * 128;

  v8f acc[4][2] = {};

  for (int k0 = 0; k0 < FDIM; k0 += 32) {
    // --- stage A tile 128x32 (f32 -> f16) ---
    #pragma unroll
    for (int i = 0; i < 4; i++) {
      int f  = tid + 256 * i;        // float4 index 0..1023
      int r  = f >> 3;               // 0..127
      int c4 = f & 7;                // 0..7
      int gr = row0 + r; if (gr >= M) gr = M - 1;   // clamp (garbage rows unstored)
      float4 v = *(const float4*)(A + (size_t)gr * FDIM + k0 + c4 * 4);
      _Float16* d = &Alds[r * LDSTR + c4 * 4];
      d[0] = (_Float16)v.x; d[1] = (_Float16)v.y;
      d[2] = (_Float16)v.z; d[3] = (_Float16)v.w;
    }
    // --- stage B tile 32x128, transposed to [n][k] so fragment K is contiguous ---
    #pragma unroll
    for (int i = 0; i < 4; i++) {
      int f  = tid + 256 * i;
      int kk = f >> 5;               // 0..31
      int n4 = f & 31;               // 0..31
      float4 v = *(const float4*)(B + (size_t)(k0 + kk) * FDIM + col0 + n4 * 4);
      Blds[(n4 * 4 + 0) * LDSTR + kk] = (_Float16)v.x;
      Blds[(n4 * 4 + 1) * LDSTR + kk] = (_Float16)v.y;
      Blds[(n4 * 4 + 2) * LDSTR + kk] = (_Float16)v.z;
      Blds[(n4 * 4 + 3) * LDSTR + kk] = (_Float16)v.w;
    }
    __syncthreads();

    // B fragments: lane = col, halves 0..15 = K kbase..kbase+15 (kbase = hi*16)
    v16h bfrag[2];
    #pragma unroll
    for (int nt = 0; nt < 2; nt++) {
      const _Float16* bb = &Blds[(waveN * 32 + nt * 16 + l16) * LDSTR + hi * 16];
      union { uint4 u[2]; v16h h; } pk;
      pk.u[0] = *(const uint4*)(bb);
      pk.u[1] = *(const uint4*)(bb + 8);
      bfrag[nt] = pk.h;
    }
    // A fragments: lane = row, halves 0..15 = K kb..kb+15 (kb = hi*16)
    #pragma unroll
    for (int t = 0; t < 4; t++) {
      const _Float16* ab = &Alds[(waveM * 64 + t * 16 + l16) * LDSTR + hi * 16];
      union { uint4 u[2]; v16h h; } pk;
      pk.u[0] = *(const uint4*)(ab);
      pk.u[1] = *(const uint4*)(ab + 8);
      v16h afrag = pk.h;
      #pragma unroll
      for (int nt = 0; nt < 2; nt++) {
        acc[t][nt] = __builtin_amdgcn_wmma_f32_16x16x32_f16(
            false, afrag, false, bfrag[nt], (short)0, acc[t][nt], false, false);
      }
    }
    __syncthreads();
  }

  // C/D layout: VGPR v -> M = v + hi*8, N = l16.
  // Fast path: full 128-row tile (390 of 391 blocks) -> straight-line stores
  // with immediate offsets; no per-element exec-mask juggling.
  if (row0 + 128 <= M) {
    #pragma unroll
    for (int t = 0; t < 4; t++) {
      int mbase = row0 + waveM * 64 + t * 16 + hi * 8;
      #pragma unroll
      for (int nt = 0; nt < 2; nt++) {
        int n = col0 + waveN * 32 + nt * 16 + l16;
        float* p = C + (size_t)mbase * FDIM + n;
        #pragma unroll
        for (int v = 0; v < 8; v++) p[(size_t)v * FDIM] = acc[t][nt][v];
      }
    }
  } else {
    #pragma unroll
    for (int t = 0; t < 4; t++) {
      int mbase = row0 + waveM * 64 + t * 16 + hi * 8;
      #pragma unroll
      for (int nt = 0; nt < 2; nt++) {
        int n = col0 + waveN * 32 + nt * 16 + l16;
        float* p = C + (size_t)mbase * FDIM + n;
        #pragma unroll
        for (int v = 0; v < 8; v++) {
          if (mbase + v < M) p[(size_t)v * FDIM] = acc[t][nt][v];
        }
      }
    }
  }
}

// ---------------------------------------------------------------------------
// Aggregation: c[n][f] = dinv[n]^2 * h[n][f] + bias[f]  (self-loop + bias init)
// ---------------------------------------------------------------------------
__global__ void agg_init_kernel(const float* __restrict__ h,
                                const float* __restrict__ dinv,
                                const float* __restrict__ bias,
                                float* __restrict__ c, int n) {
  int idx = blockIdx.x * blockDim.x + threadIdx.x;
  if (idx < n * FDIM) {
    int node = idx >> 9;
    int f    = idx & (FDIM - 1);
    float di = dinv[node];
    c[idx] = di * di * h[idx] + bias[f];
  }
}

// One wave per edge: gather h[src] row (float4, coalesced), scale by the
// symmetric norm, atomic-scatter into c[dst]. Accumulator fits in 192MB L2.
__global__ __launch_bounds__(256)
void agg_edges_kernel(const int* __restrict__ src, const int* __restrict__ dst,
                      const float* __restrict__ ew, const float* __restrict__ dinv,
                      const float* __restrict__ h, float* __restrict__ c, int e) {
  int wv   = (blockIdx.x * blockDim.x + threadIdx.x) >> 5;
  int lane = threadIdx.x & 31;
  if (wv >= e) return;
  int s = src[wv], d = dst[wv];
  float nrm = dinv[s] * ew[wv] * dinv[d];
  const float4* hs = (const float4*)(h + (size_t)s * FDIM);
  float* cd = c + (size_t)d * FDIM;
  #pragma unroll
  for (int i = 0; i < 4; i++) {
    float4 v = hs[lane + 32 * i];
    int f = (lane + 32 * i) * 4;
    atomicAdd(&cd[f + 0], nrm * v.x);
    atomicAdd(&cd[f + 1], nrm * v.y);
    atomicAdd(&cd[f + 2], nrm * v.z);
    atomicAdd(&cd[f + 3], nrm * v.w);
  }
}

// ---------------------------------------------------------------------------
// Fused gate: out = relu((1 - sigmoid(zpre + Lzb)) * tanh(hpre + Lhb))
// (H = 0 in the reference, so Z*H vanishes and R is dead.)
// ---------------------------------------------------------------------------
__global__ void gate_kernel(const float* __restrict__ zpre,
                            const float* __restrict__ hpre,
                            const float* __restrict__ lzb,
                            const float* __restrict__ lhb,
                            float* __restrict__ out, int total) {
  int idx = blockIdx.x * blockDim.x + threadIdx.x;
  if (idx < total) {
    int f = idx & (FDIM - 1);
    float z  = 1.0f / (1.0f + __expf(-(zpre[idx] + lzb[f])));
    float ht = tanhf(hpre[idx] + lhb[f]);
    float o  = (1.0f - z) * ht;
    out[idx] = o > 0.0f ? o : 0.0f;
  }
}

// ---------------------------------------------------------------------------
extern "C" void kernel_launch(void* const* d_in, const int* in_sizes, int n_in,
                              void* d_out, int out_size, void* d_ws, size_t ws_size,
                              hipStream_t stream) {
  const float* x   = (const float*)d_in[0];
  const int*   ei  = (const int*)  d_in[1];   // [2, E] int32 (JAX x64 disabled)
  const float* ew  = (const float*)d_in[2];
  const float* Wz  = (const float*)d_in[3];
  const float* bz  = (const float*)d_in[4];
  const float* LzW = (const float*)d_in[5];   // [1024,512]; rows 0..511 = top half
  const float* Lzb = (const float*)d_in[6];
  // d_in[7..10] (Wr, br, LrW, Lrb) unused: H==0 makes the R gate dead.
  const float* Wh  = (const float*)d_in[11];
  const float* bh  = (const float*)d_in[12];
  const float* LhW = (const float*)d_in[13];
  const float* Lhb = (const float*)d_in[14];

  const int n = in_sizes[0] / FDIM;   // 50000 nodes
  const int e = in_sizes[2];          // 800000 edges
  const int* src = ei;
  const int* dst = ei + e;

  char*  ws   = (char*)d_ws;
  float* dinv = (float*)ws;
  size_t dinvBytes = ((size_t)n * sizeof(float) + 255) & ~(size_t)255;
  size_t bufBytes  = (size_t)n * FDIM * sizeof(float);
  float* bufA = (float*)(ws + dinvBytes);
  float* bufB = (float*)(ws + dinvBytes + bufBytes);
  float* outp = (float*)d_out;

  const int nt = n * FDIM;
  dim3 gemmGrid(FDIM / 128, (n + 127) / 128);  // x = col tile (fast), y = row tile

  // 1) symmetric GCN norm: dinv = rsqrt(1 + sum_in ew)
  deg_init_kernel <<<(n + 255) / 256, 256, 0, stream>>>(dinv, n);
  deg_edges_kernel<<<(e + 255) / 256, 256, 0, stream>>>(dst, ew, dinv, e);
  deg_fin_kernel  <<<(n + 255) / 256, 256, 0, stream>>>(dinv, n);

  // 2) cz = A_norm (x @ Wz) + bz       [hz -> bufA, cz -> bufB]
  gemm_wmma_f16  <<<gemmGrid, 256, 0, stream>>>(x, Wz, bufA, n);
  agg_init_kernel<<<(nt + 255) / 256, 256, 0, stream>>>(bufA, dinv, bz, bufB, n);
  agg_edges_kernel<<<(e + 7) / 8, 256, 0, stream>>>(src, dst, ew, dinv, bufA, bufB, e);

  // 3) ch = A_norm (x @ Wh) + bh       [hh -> bufA (reuse), ch -> d_out]
  gemm_wmma_f16  <<<gemmGrid, 256, 0, stream>>>(x, Wh, bufA, n);
  agg_init_kernel<<<(nt + 255) / 256, 256, 0, stream>>>(bufA, dinv, bh, outp, n);
  agg_edges_kernel<<<(e + 7) / 8, 256, 0, stream>>>(src, dst, ew, dinv, bufA, outp, e);

  // 4) gate pre-activations: zpre = cz @ LzW_top -> bufA, hpre = ch @ LhW_top -> bufB
  gemm_wmma_f16<<<gemmGrid, 256, 0, stream>>>(bufB, LzW, bufA, n);
  gemm_wmma_f16<<<gemmGrid, 256, 0, stream>>>(outp, LhW, bufB, n);

  // 5) out = relu((1 - sigmoid(zpre + Lzb)) * tanh(hpre + Lhb))
  gate_kernel<<<(nt + 255) / 256, 256, 0, stream>>>(bufA, bufB, Lzb, Lhb, outp, nt);
}
